// MeanShift_32469952758150
// MI455X (gfx1250) — matrix operand order repaired and checked by reading, not compiled, not run on hardware
//
#include <hip/hip_runtime.h>
#include <hip/hip_bf16.h>

// ---------------------------------------------------------------------------
// Problem constants (fixed by the reference setup)
// ---------------------------------------------------------------------------
constexpr int B_DIM = 256;     // rows (batch)
constexpr int C_DIM = 512;     // feature dim (GEMM K-reduction)
constexpr int K_DIM = 128000;  // queue entries (GEMM N)
constexpr int TOPK  = 5;

// GEMM/top-k kernel tiling
constexpr int COLS    = 32;                          // queue columns per iteration
constexpr int ITERS   = 8;                           // iterations per block
constexpr int NBLK    = K_DIM / (COLS * ITERS);      // 500 blocks
constexpr int QSTRIDE = C_DIM + 8;                   // bf16 LDS stride (pad vs bank conflicts)
constexpr int DSTRIDE = COLS + 4;                    // dist tile stride (pad)

typedef __attribute__((ext_vector_type(16))) __bf16 v16bf;
typedef __attribute__((ext_vector_type(8)))  float  v8f;

struct U32x8 { uint4 a, b; };   // 32 bytes, bit-cast carrier for v16bf

// A-fragment: 16-bit A 16x32 layout. p already points at row base + half*8 + c0.
// elements [0..7] = K(c0+half*8 .. +7), elements [8..15] = same +16.
__device__ __forceinline__ v16bf load_afrag(const __bf16* p) {
    U32x8 u;
    u.a = *(const uint4*)(p);
    u.b = *(const uint4*)(p + 16);
    return __builtin_bit_cast(v16bf, u);
}

// B-fragment: 16-bit B 32x16 layout, same K order as A: p points at
// qtile + col*QSTRIDE + c0 + half*8; elements [0..7] = K(half*8..+7), [8..15] = same +16.
__device__ __forceinline__ v16bf load_bfrag(const __bf16* p) {
    U32x8 u;
    u.a = *(const uint4*)(p);
    u.b = *(const uint4*)(p + 16);
    return __builtin_bit_cast(v16bf, u);
}

// Keep 5 smallest (key asc). Fully unrolled -> stays in VGPRs (v_cndmask).
__device__ __forceinline__ void push_min(float d, float p, float nd[TOPK], float np[TOPK]) {
    if (d < nd[4]) {
        nd[4] = d; np[4] = p;
#pragma unroll
        for (int j = 4; j > 0; --j) {
            if (nd[j] < nd[j - 1]) {
                float t0 = nd[j]; nd[j] = nd[j - 1]; nd[j - 1] = t0;
                float t1 = np[j]; np[j] = np[j - 1]; np[j - 1] = t1;
            }
        }
    }
}

// Keep 5 largest (key desc).
__device__ __forceinline__ void push_max(float d, float p, float fd[TOPK], float fp[TOPK]) {
    if (d > fd[4]) {
        fd[4] = d; fp[4] = p;
#pragma unroll
        for (int j = 4; j > 0; --j) {
            if (fd[j] > fd[j - 1]) {
                float t0 = fd[j]; fd[j] = fd[j - 1]; fd[j - 1] = t0;
                float t1 = fp[j]; fp[j] = fp[j - 1]; fp[j - 1] = t1;
            }
        }
    }
}

// ---------------------------------------------------------------------------
// Kernel 1: L2-normalize rows of query/target, emit bf16 row-major.
// grid = (256 rows, 2 tensors), block = 256
// ---------------------------------------------------------------------------
__global__ void nrm_kernel(const float* __restrict__ query,
                           const float* __restrict__ target,
                           __bf16* __restrict__ qn,
                           __bf16* __restrict__ tn) {
    __shared__ float red[256];
    const int row = blockIdx.x;
    const int t   = threadIdx.x;
    const float* src = blockIdx.y ? query : target;
    __bf16*      dst = blockIdx.y ? qn    : tn;

    const float x0 = src[(size_t)row * C_DIM + t];
    const float x1 = src[(size_t)row * C_DIM + 256 + t];
    red[t] = x0 * x0 + x1 * x1;
    __syncthreads();
#pragma unroll
    for (int s = 128; s > 0; s >>= 1) {
        if (t < s) red[t] += red[t + s];
        __syncthreads();
    }
    const float rn = 1.0f / fmaxf(sqrtf(red[0]), 1e-12f);
    dst[(size_t)row * C_DIM + t]       = (__bf16)(x0 * rn);
    dst[(size_t)row * C_DIM + 256 + t] = (__bf16)(x1 * rn);
}

// ---------------------------------------------------------------------------
// Kernel 2: fused dual-GEMM (bf16 WMMA) + streaming per-row top-5 near/far.
// Each block owns a K-strip of COLS*ITERS queue columns and ALL 256 rows, so
// the 262 MB queue is streamed from HBM exactly once.
// Per wave: rt in {wave, wave+8}, both 16-col tiles accumulated together so
// each A-fragment pair feeds 4 WMMAs (A reuse x2, B reuse x2) -> bounded
// register pressure, no spills.
// ---------------------------------------------------------------------------
__global__ __launch_bounds__(256, 1)
void gemm_topk_kernel(const float*  __restrict__ queue,
                      const __bf16* __restrict__ tn,
                      const __bf16* __restrict__ qn,
                      float* __restrict__ candN,
                      float* __restrict__ candF) {
    extern __shared__ char smem[];
    __bf16* qtile = (__bf16*)smem;                                    // 32*520*2  = 33280 B
    float*  sdt   = (float*)(smem + COLS * QSTRIDE * sizeof(__bf16)); // 256*36*4  = 36864 B
    float*  sdq   = sdt + B_DIM * DSTRIDE;                            // 256*36*4  = 36864 B

    const int tid  = threadIdx.x;
    const int lane = tid & 31;
    const int wave = tid >> 5;
    const int half = lane >> 4;   // 0 | 1
    const int l15  = lane & 15;

    // Per-thread running top-5 for row == tid (key = dist_t, payload = dist_q)
    float nd[TOPK], nq[TOPK], fd[TOPK], fq[TOPK];
#pragma unroll
    for (int i = 0; i < TOPK; ++i) {
        nd[i] = __builtin_inff();  nq[i] = 0.0f;
        fd[i] = -__builtin_inff(); fq[i] = 0.0f;
    }

#pragma unroll 1
    for (int it = 0; it < ITERS; ++it) {
        const int colBase = (blockIdx.x * ITERS + it) * COLS;

        // -------- Phase A: stage 32 queue rows (fp32 -> bf16) into LDS -----
        __syncthreads();  // previous iteration fully consumed LDS
#pragma unroll
        for (int j = 0; j < 16; ++j) {
            const int flat = j * 1024 + tid * 4;    // fully coalesced float4
            const int colL = flat >> 9;             // 0..31
            const int c    = flat & 511;
            const float4 f = *(const float4*)(queue + (size_t)(colBase + colL) * C_DIM + c);
            union { __bf16 h[4]; uint2 u; } pk;
            pk.h[0] = (__bf16)f.x; pk.h[1] = (__bf16)f.y;
            pk.h[2] = (__bf16)f.z; pk.h[3] = (__bf16)f.w;
            *(uint2*)(qtile + colL * QSTRIDE + c) = pk.u;
        }
        __syncthreads();

        // -------- Phase B: WMMA dual-GEMM, 256 rows x 32 cols --------------
#pragma unroll 1
        for (int g = 0; g < 2; ++g) {
            const int rt   = wave + g * 8;     // 8 waves x 2 groups = 16 row-tiles
            const int row0 = rt * 16;

            v8f acct0 = {};   // dist_t tile, cols [0,16)
            v8f acct1 = {};   // dist_t tile, cols [16,32)
            v8f accq0 = {};   // dist_q tile, cols [0,16)
            v8f accq1 = {};   // dist_q tile, cols [16,32)

            const __bf16* tA  = tn + (size_t)(row0 + l15) * C_DIM + half * 8;
            const __bf16* qA  = qn + (size_t)(row0 + l15) * C_DIM + half * 8;
            const __bf16* bB0 = qtile + l15        * QSTRIDE + half * 8;
            const __bf16* bB1 = qtile + (16 + l15) * QSTRIDE + half * 8;

#pragma unroll 2
            for (int cs = 0; cs < 16; ++cs) {
                const int c0 = cs * 32;
                const v16bf at  = load_afrag(tA + c0);
                const v16bf aq  = load_afrag(qA + c0);
                const v16bf bb0 = load_bfrag(bB0 + c0);
                const v16bf bb1 = load_bfrag(bB1 + c0);
                acct0 = __builtin_amdgcn_wmma_f32_16x16x32_bf16(
                            false, at, false, bb0, (short)0, acct0, false, false);
                accq0 = __builtin_amdgcn_wmma_f32_16x16x32_bf16(
                            false, aq, false, bb0, (short)0, accq0, false, false);
                acct1 = __builtin_amdgcn_wmma_f32_16x16x32_bf16(
                            false, at, false, bb1, (short)0, acct1, false, false);
                accq1 = __builtin_amdgcn_wmma_f32_16x16x32_bf16(
                            false, aq, false, bb1, (short)0, accq1, false, false);
            }
            // C/D layout: VGPR v, lane<16 -> M=v, lane>=16 -> M=v+8; N = lane&15
#pragma unroll
            for (int v = 0; v < 8; ++v) {
                const int r  = row0 + v + half * 8;
                const int c0 = l15;
                const int c1 = 16 + l15;
                sdt[r * DSTRIDE + c0] = 2.0f - 2.0f * acct0[v];
                sdt[r * DSTRIDE + c1] = 2.0f - 2.0f * acct1[v];
                sdq[r * DSTRIDE + c0] = 2.0f - 2.0f * accq0[v];
                sdq[r * DSTRIDE + c1] = 2.0f - 2.0f * accq1[v];
            }
        }
        __syncthreads();

        // -------- Phase C: thread tid scans its row's 32 new columns -------
#pragma unroll 4
        for (int c = 0; c < COLS; ++c) {
            const float dt = sdt[tid * DSTRIDE + c];
            const float dq = sdq[tid * DSTRIDE + c];
            push_min(dt, dq, nd, nq);
            push_max(dt, dq, fd, fq);
        }
    }

    // -------- emit per-(row, block) candidate lists ------------------------
    float* oN = candN + ((size_t)tid * NBLK + blockIdx.x) * (2 * TOPK);
    float* oF = candF + ((size_t)tid * NBLK + blockIdx.x) * (2 * TOPK);
#pragma unroll
    for (int i = 0; i < TOPK; ++i) {
        oN[2 * i] = nd[i]; oN[2 * i + 1] = nq[i];
        oF[2 * i] = fd[i]; oF[2 * i + 1] = fq[i];
    }
}

// ---------------------------------------------------------------------------
// Kernel 3: merge candidates per row, reduce to scalar loss. 1 block x 256.
// ---------------------------------------------------------------------------
__global__ void merge_kernel(const float* __restrict__ candN,
                             const float* __restrict__ candF,
                             const int* __restrict__ ep,
                             const int* __restrict__ eps,
                             const int* __restrict__ tk,
                             float* __restrict__ out) {
    __shared__ float sn[256], sf[256];
    const int b = threadIdx.x;

    float nd[TOPK], nq[TOPK], fd[TOPK], fq[TOPK];
#pragma unroll
    for (int i = 0; i < TOPK; ++i) {
        nd[i] = __builtin_inff();  nq[i] = 0.0f;
        fd[i] = -__builtin_inff(); fq[i] = 0.0f;
    }

    const float* pN = candN + (size_t)b * NBLK * (2 * TOPK);
    const float* pF = candF + (size_t)b * NBLK * (2 * TOPK);
    for (int i = 0; i < NBLK * TOPK; ++i) {
        push_min(pN[2 * i], pN[2 * i + 1], nd, nq);
        push_max(pF[2 * i], pF[2 * i + 1], fd, fq);
    }
    float sumn = 0.0f, sumf = 0.0f;
#pragma unroll
    for (int i = 0; i < TOPK; ++i) { sumn += nq[i]; sumf += fq[i]; }

    const float invk = 1.0f / (float)tk[0];
    sn[b] = sumn * invk;
    sf[b] = sumf * invk;
    __syncthreads();
#pragma unroll
    for (int s = 128; s > 0; s >>= 1) {
        if (b < s) { sn[b] += sn[b + s]; sf[b] += sf[b + s]; }
        __syncthreads();
    }
    if (b == 0) {
        const float near_term = sn[0] * (1.0f / (float)B_DIM);
        const float far_term  = sf[0] * (1.0f / (float)B_DIM);
        const int e = ep[0], E = eps[0];
        float loss = near_term;
        if (2 * e < E) loss -= ((float)(E - e) / (float)E) * far_term;
        out[0] = loss;
    }
}

// ---------------------------------------------------------------------------
extern "C" void kernel_launch(void* const* d_in, const int* in_sizes, int n_in,
                              void* d_out, int out_size, void* d_ws, size_t ws_size,
                              hipStream_t stream) {
    const float* query  = (const float*)d_in[0];
    const float* target = (const float*)d_in[1];
    const float* queue  = (const float*)d_in[2];
    const int*   epoch  = (const int*)d_in[3];
    const int*   epochs = (const int*)d_in[4];
    const int*   topk   = (const int*)d_in[5];
    float*       out    = (float*)d_out;

    // workspace layout
    char* ws = (char*)d_ws;
    __bf16* tn    = (__bf16*)ws;                               // 256*512*2 = 262144 B
    __bf16* qn    = (__bf16*)(ws + 262144);                    // 262144 B
    float*  candN = (float*)(ws + 524288);                     // 256*500*10*4 = 5,120,000 B
    float*  candF = (float*)(ws + 524288 + 5120000);           // 5,120,000 B

    nrm_kernel<<<dim3(B_DIM, 2), 256, 0, stream>>>(query, target, qn, tn);

    const size_t shmem = (size_t)COLS * QSTRIDE * sizeof(__bf16)
                       + (size_t)2 * B_DIM * DSTRIDE * sizeof(float);   // 107,008 B
    hipFuncSetAttribute((const void*)gemm_topk_kernel,
                        hipFuncAttributeMaxDynamicSharedMemorySize, (int)shmem);
    gemm_topk_kernel<<<dim3(NBLK), 256, shmem, stream>>>(queue, tn, qn, candN, candF);

    merge_kernel<<<1, 256, 0, stream>>>(candN, candF, epoch, epochs, topk, out);
}
